// LoRACrossAttention_38817914421821
// MI455X (gfx1250) — matrix operand order repaired and checked by reading, not compiled, not run on hardware
//
#include <hip/hip_runtime.h>
#include <cstdint>

// Problem constants (match reference)
#define BB 2
#define NN 2048
#define MM 2048
#define DD 1024
#define HH 16
#define DHH 64
#define INNER 1024
#define RR 16
#define SCALE 0.125f

typedef __attribute__((ext_vector_type(16))) __bf16 v16bf;
typedef __attribute__((ext_vector_type(8)))  float  v8f;
typedef __attribute__((ext_vector_type(4)))  float  v4f;
typedef __attribute__((ext_vector_type(4)))  __bf16 v4bf;
typedef __attribute__((__vector_size__(4 * sizeof(int)))) int v4i;

union Frag16 { v16bf v; unsigned u[8]; };

// Native bf16 converts (backend emits v_cvt_pk_bf16_f32)
__device__ __forceinline__ unsigned short f2bf(float f) {
  union { __bf16 h; unsigned short u; } c;
  c.h = (__bf16)f;
  return c.u;
}
// Convert 4 floats -> 4 bf16 in one vector fptrunc (2x v_cvt_pk_bf16_f32, no fixup)
__device__ __forceinline__ uint2 pack4_bf16(float4 f) {
  v4f vf = {f.x, f.y, f.z, f.w};
  v4bf vb = __builtin_convertvector(vf, v4bf);
  union { v4bf b; uint2 u; } c;
  c.b = vb;
  return c.u;
}

// ---- CDNA5 async global->LDS path (ASYNCcnt) ------------------------------
#define GLOBAL_AS __attribute__((address_space(1)))
#define LDS_AS    __attribute__((address_space(3)))

#if __has_builtin(__builtin_amdgcn_global_load_async_to_lds_b128)
#define HAVE_ASYNC 1
__device__ __forceinline__ void async_b128(const void* g, void* l) {
  __builtin_amdgcn_global_load_async_to_lds_b128(
      (GLOBAL_AS v4i*)(uintptr_t)g,
      (LDS_AS v4i*)(unsigned)(uintptr_t)l,
      0, 0);
}
__device__ __forceinline__ void wait_async() {
  asm volatile("s_wait_asynccnt 0x0" ::: "memory");
}
#else
#define HAVE_ASYNC 0
#endif

__device__ __forceinline__ v8f wmma_bf16(v16bf a, v16bf b, v8f c) {
  // (neg_a, A, neg_b, B, c_mod, C, reuse_a, reuse_b)
  return __builtin_amdgcn_wmma_f32_16x16x32_bf16(false, a, false, b, (short)0, c, false, false);
}

// A-matrix 16x32 bf16 fragment (ISA 7.12.2): lane m=l&15, half=l>>4.
// VGPR v<4 : K = 8*half + 2v ; v>=4 : K = 16 + 8*half + 2(v-4).
__device__ __forceinline__ v16bf load_fragA(const unsigned short* base, int row0, int pitch) {
  const int l = threadIdx.x & 31;
  const int m = l & 15, hw = l >> 4;
  const unsigned short* rp = base + (long)(row0 + m) * pitch;
  Frag16 f;
#pragma unroll
  for (int vv = 0; vv < 8; ++vv) {
    int kb = (vv < 4) ? (8 * hw + 2 * vv) : (16 + 8 * hw + 2 * (vv - 4));
    f.u[vv] = *(const unsigned*)(rp + kb);
  }
  return f.v;
}

// B-matrix 32x16 bf16 fragment: lane n=l&15, half=l>>4. VGPR v<4 : K = 8*half + 2v ; v>=4 : K = 16 + 8*half + 2(v-4).
// Source layout must be [n][k] row-major (k contiguous).
__device__ __forceinline__ v16bf load_fragB(const unsigned short* base, int row0, int pitch) {
  const int l = threadIdx.x & 31;
  const int n = l & 15, hw = l >> 4;
  const unsigned short* rp = base + (long)(row0 + n) * pitch;
  Frag16 f;
#pragma unroll
  for (int vv = 0; vv < 8; ++vv) {
    int kb = (vv < 4) ? (8 * hw + 2 * vv) : (16 + 8 * hw + 2 * (vv - 4));
    f.u[vv] = *(const unsigned*)(rp + kb);
  }
  return f.v;
}

// ---------------------------------------------------------------------------
// 1) W_eff[b] = W + B[task[b]] @ A[task[b]]   (rank-16 fold)
// grid: (INNER*DD/256, BB, 2)   z: 0 -> K weights, 1 -> V weights
// ---------------------------------------------------------------------------
__global__ __launch_bounds__(256) void build_weff(
    const float* __restrict__ Wk, const float* __restrict__ Ak, const float* __restrict__ Bk,
    const float* __restrict__ Wv, const float* __restrict__ Av, const float* __restrict__ Bv,
    const int* __restrict__ task, float* __restrict__ WkE, float* __restrict__ WvE)
{
  const int which = blockIdx.z;
  const float* W = which ? Wv : Wk;
  const float* A = which ? Av : Ak;
  const float* Bm = which ? Bv : Bk;
  float* E = which ? WvE : WkE;
  const int b = blockIdx.y;
  const int t = task[b];
  const long e = (long)blockIdx.x * 256 + threadIdx.x;  // e in [0, INNER*DD)
  const int i = (int)(e >> 10);
  const int d = (int)(e & 1023);
  float s = W[e];
  const float* Ab = A + (long)t * RR * DD + d;           // stride DD per r
  const float* Bb = Bm + (long)t * INNER * RR + (long)i * RR;
#pragma unroll
  for (int r = 0; r < RR; ++r) s += Bb[r] * Ab[(long)r * DD];
  E[(long)b * INNER * DD + e] = s;
}

// ---------------------------------------------------------------------------
// 2) C_bf16[m][n] = A_f32[m][k] @ Bt_f32[n][k]^T
// WG tile 128x128, 8 waves in 4x2, each wave 32x64 (8 WMMAs / K-step)
// grid: (Mrows/128, Ncols/128, batch)
// ---------------------------------------------------------------------------
__global__ __launch_bounds__(256) void gemm_f32_bf16out(
    const float* __restrict__ A, long sAz,
    const float* __restrict__ Bt, long sBz,
    unsigned short* __restrict__ C, long sCz,
    int K, int Nc)
{
  __shared__ unsigned short sA[128 * 32];
  __shared__ unsigned short sB[128 * 32];
  const int tid = threadIdx.x;
  const int wid = tid >> 5;
  const int wm = wid >> 1, wn = wid & 1;
  const int lda = K, ldb = K, ldc = Nc;
  const float* Ab = A + (long)blockIdx.z * sAz + (long)(blockIdx.x * 128) * lda;
  const float* Bb = Bt + (long)blockIdx.z * sBz + (long)(blockIdx.y * 128) * ldb;

  v8f acc[2][4];
#pragma unroll
  for (int i = 0; i < 2; ++i)
#pragma unroll
    for (int j = 0; j < 4; ++j) {
      v8f z = {0, 0, 0, 0, 0, 0, 0, 0};
      acc[i][j] = z;
    }

  for (int k0 = 0; k0 < K; k0 += 32) {
    // stage A and B tiles: 128x32 f32 -> bf16 (float4 loads, packed cvt, 8B LDS stores)
#pragma unroll
    for (int e = tid; e < 128 * 8; e += 256) {
      int row = e >> 3, c4 = e & 7;
      float4 f = *(const float4*)(Ab + (long)row * lda + k0 + c4 * 4);
      *(uint2*)(&sA[row * 32 + c4 * 4]) = pack4_bf16(f);
    }
#pragma unroll
    for (int e = tid; e < 128 * 8; e += 256) {
      int row = e >> 3, c4 = e & 7;
      float4 f = *(const float4*)(Bb + (long)row * ldb + k0 + c4 * 4);
      *(uint2*)(&sB[row * 32 + c4 * 4]) = pack4_bf16(f);
    }
    __syncthreads();

    if (k0 + 32 < K && tid < 128)  // prefetch next A slab -> global_prefetch_b8
      __builtin_prefetch(Ab + (long)tid * lda + k0 + 32, 0, 1);

    v16bf a0 = load_fragA(sA, wm * 32, 32);
    v16bf a1 = load_fragA(sA, wm * 32 + 16, 32);
    v16bf bf[4];
#pragma unroll
    for (int j = 0; j < 4; ++j) bf[j] = load_fragB(sB, wn * 64 + 16 * j, 32);
#pragma unroll
    for (int j = 0; j < 4; ++j) {
      acc[0][j] = wmma_bf16(a0, bf[j], acc[0][j]);
      acc[1][j] = wmma_bf16(a1, bf[j], acc[1][j]);
    }
    __syncthreads();
  }

  const int l = tid & 31, lane_n = l & 15, hw = l >> 4;
  unsigned short* Cb = C + (long)blockIdx.z * sCz;
  const int row0 = blockIdx.x * 128 + wm * 32 + 8 * hw;
  const int colb = blockIdx.y * 128 + wn * 64 + lane_n;
#pragma unroll
  for (int i = 0; i < 2; ++i)
#pragma unroll
    for (int j = 0; j < 4; ++j)
#pragma unroll
      for (int r = 0; r < 8; ++r)
        Cb[(long)(row0 + 16 * i + r) * ldc + colb + 16 * j] = f2bf(acc[i][j][r]);
}

// ---------------------------------------------------------------------------
// 3) Flash attention: one WG per (b, h, 128-query slab); 8 waves x 16 queries
// ---------------------------------------------------------------------------
__global__ __launch_bounds__(256) void attn_kernel(
    const unsigned short* __restrict__ q,
    const unsigned short* __restrict__ kk,
    const unsigned short* __restrict__ vptr,
    const unsigned char* __restrict__ mask,
    unsigned short* __restrict__ o)
{
  __shared__ unsigned short sK[32 * 64];     // [ctx][d]
  __shared__ unsigned short sV[64 * 32];     // [d][ctx] (transposed)
  __shared__ unsigned short sP[8][16 * 32];  // per-wave P round-trip
  const int b = blockIdx.z, h = blockIdx.y;
  const int tid = threadIdx.x, wid = tid >> 5, l = tid & 31;
  const int lane_n = l & 15, hw = l >> 4;
  const int q_base = blockIdx.x * 128 + wid * 16;

  // Q fragments (load-once, direct from global bf16, natural [m][k] layout)
  const unsigned short* qb = q + ((long)(b * NN + q_base)) * INNER + h * DHH;
  v16bf fa0 = load_fragA(qb, 0, INNER);
  v16bf fa1 = load_fragA(qb + 32, 0, INNER);

  float row_m[8], row_l[8];
  v8f z = {0, 0, 0, 0, 0, 0, 0, 0};
  v8f acc0 = z, acc1 = z, acc2 = z, acc3 = z;
#pragma unroll
  for (int r = 0; r < 8; ++r) { row_m[r] = -3.0e38f; row_l[r] = 0.f; }

  for (int c0 = 0; c0 < MM; c0 += 32) {
    // stage K: 32 ctx rows x 64 d
#if HAVE_ASYNC
    {
      int row = tid >> 3, c8 = tid & 7;  // 32 rows x 8 chunks of 16B
      async_b128(kk + ((long)(b * MM + c0 + row)) * INNER + h * DHH + c8 * 8,
                 &sK[row * 64 + c8 * 8]);
    }
#else
#pragma unroll
    for (int e = tid; e < 1024; e += 256) {
      int row = e >> 5, c2 = e & 31;
      ((unsigned*)sK)[e] =
          *(const unsigned*)(kk + ((long)(b * MM + c0 + row)) * INNER + h * DHH + c2 * 2);
    }
#endif
    // stage V transposed: [d][ctx]
#pragma unroll
    for (int e = tid; e < 1024; e += 256) {
      int row = e >> 5, c2 = e & 31;
      unsigned u = *(const unsigned*)(vptr + ((long)(b * MM + c0 + row)) * INNER + h * DHH + c2 * 2);
      sV[(c2 * 2) * 32 + row]     = (unsigned short)(u & 0xFFFFu);
      sV[(c2 * 2 + 1) * 32 + row] = (unsigned short)(u >> 16);
    }
#if HAVE_ASYNC
    wait_async();
#endif
    __syncthreads();

    // sim = Q(16x64) . K^T : two 16-col tiles, K-dim 64 = 2 WMMAs each
    v8f s0 = wmma_bf16(fa0, load_fragB(sK, 0, 64), z);
    s0     = wmma_bf16(fa1, load_fragB(sK + 32, 0, 64), s0);
    v8f s1 = wmma_bf16(fa0, load_fragB(sK, 16, 64), z);
    s1     = wmma_bf16(fa1, load_fragB(sK + 32, 16, 64), s1);

    const unsigned char mk0 = mask[(long)b * MM + c0 + lane_n];
    const unsigned char mk1 = mask[(long)b * MM + c0 + 16 + lane_n];
    const float neg = -3.0e38f;

    float p0[8], p1[8];
#pragma unroll
    for (int r = 0; r < 8; ++r) {
      float x0 = mk0 ? s0[r] * SCALE : neg;
      float x1 = mk1 ? s1[r] * SCALE : neg;
      float mx = fmaxf(x0, x1);
#pragma unroll
      for (int off = 8; off >= 1; off >>= 1) mx = fmaxf(mx, __shfl_xor(mx, off, 16));
      float nm = fmaxf(row_m[r], mx);
      float corr = __expf(row_m[r] - nm);
      float e0 = __expf(x0 - nm), e1 = __expf(x1 - nm);
      float sum = e0 + e1;
#pragma unroll
      for (int off = 8; off >= 1; off >>= 1) sum += __shfl_xor(sum, off, 16);
      row_l[r] = row_l[r] * corr + sum;
      row_m[r] = nm;
      p0[r] = e0; p1[r] = e1;
      acc0[r] *= corr; acc1[r] *= corr; acc2[r] *= corr; acc3[r] *= corr;
    }

    // C-layout -> A-fragment layout via per-wave LDS round trip
    unsigned short* myP = sP[wid];
#pragma unroll
    for (int r = 0; r < 8; ++r) {
      int row = r + 8 * hw;
      myP[row * 32 + lane_n]      = f2bf(p0[r]);
      myP[row * 32 + 16 + lane_n] = f2bf(p1[r]);
    }
    __builtin_amdgcn_wave_barrier();  // keep cross-lane LDS store->load ordered
    v16bf pa = load_fragA(myP, 0, 32);

    acc0 = wmma_bf16(pa, load_fragB(sV, 0, 32), acc0);
    acc1 = wmma_bf16(pa, load_fragB(sV, 16, 32), acc1);
    acc2 = wmma_bf16(pa, load_fragB(sV, 32, 32), acc2);
    acc3 = wmma_bf16(pa, load_fragB(sV, 48, 32), acc3);
    __syncthreads();
  }

#pragma unroll
  for (int r = 0; r < 8; ++r) {
    int row_g = q_base + r + 8 * hw;
    float inv = 1.0f / row_l[r];
    long base = ((long)(b * NN + row_g)) * INNER + h * DHH + lane_n;
    o[base]      = f2bf(acc0[r] * inv);
    o[base + 16] = f2bf(acc1[r] * inv);
    o[base + 32] = f2bf(acc2[r] * inv);
    o[base + 48] = f2bf(acc3[r] * inv);
  }
}

// ---------------------------------------------------------------------------
// 4) out_f32[m][n] = A_bf16[m][k] @ Wo_f32[n][k]^T + bo[n]
// WG tile 128x128, wave tile 32x64
// ---------------------------------------------------------------------------
__global__ __launch_bounds__(256) void gemm_bf16_f32bias(
    const unsigned short* __restrict__ A, long sAz,
    const float* __restrict__ Bt,
    float* __restrict__ C, long sCz,
    const float* __restrict__ bo,
    int K, int Nc)
{
  __shared__ unsigned short sA[128 * 32];
  __shared__ unsigned short sB[128 * 32];
  const int tid = threadIdx.x;
  const int wid = tid >> 5;
  const int wm = wid >> 1, wn = wid & 1;
  const int lda = K, ldb = K, ldc = Nc;
  const unsigned short* Ab = A + (long)blockIdx.z * sAz + (long)(blockIdx.x * 128) * lda;
  const float* Bb = Bt + (long)(blockIdx.y * 128) * ldb;

  v8f acc[2][4];
#pragma unroll
  for (int i = 0; i < 2; ++i)
#pragma unroll
    for (int j = 0; j < 4; ++j) {
      v8f z = {0, 0, 0, 0, 0, 0, 0, 0};
      acc[i][j] = z;
    }

  for (int k0 = 0; k0 < K; k0 += 32) {
    // A bf16 tile: raw 16B copies (async to LDS if available)
#if HAVE_ASYNC
#pragma unroll
    for (int e = tid; e < 512; e += 256) {  // 128 rows x 4 chunks of 8 ushorts
      int row = e >> 2, c8 = e & 3;
      async_b128(Ab + (long)row * lda + k0 + c8 * 8, &sA[row * 32 + c8 * 8]);
    }
#else
#pragma unroll
    for (int e = tid; e < 2048; e += 256) {
      int row = e >> 4, c2 = e & 15;
      ((unsigned*)sA)[row * 16 + c2] =
          *(const unsigned*)(Ab + (long)row * lda + k0 + c2 * 2);
    }
#endif
#pragma unroll
    for (int e = tid; e < 128 * 8; e += 256) {
      int row = e >> 3, c4 = e & 7;
      float4 f = *(const float4*)(Bb + (long)row * ldb + k0 + c4 * 4);
      *(uint2*)(&sB[row * 32 + c4 * 4]) = pack4_bf16(f);
    }
#if HAVE_ASYNC
    wait_async();
#endif
    __syncthreads();

    v16bf a0 = load_fragA(sA, wm * 32, 32);
    v16bf a1 = load_fragA(sA, wm * 32 + 16, 32);
    v16bf bf[4];
#pragma unroll
    for (int j = 0; j < 4; ++j) bf[j] = load_fragB(sB, wn * 64 + 16 * j, 32);
#pragma unroll
    for (int j = 0; j < 4; ++j) {
      acc[0][j] = wmma_bf16(a0, bf[j], acc[0][j]);
      acc[1][j] = wmma_bf16(a1, bf[j], acc[1][j]);
    }
    __syncthreads();
  }

  const int l = tid & 31, lane_n = l & 15, hw = l >> 4;
  float* Cb = C + (long)blockIdx.z * sCz;
  const int row0 = blockIdx.x * 128 + wm * 32 + 8 * hw;
  const int colb = blockIdx.y * 128 + wn * 64 + lane_n;
#pragma unroll
  for (int j = 0; j < 4; ++j) {
    const float bias = bo[colb + 16 * j];
#pragma unroll
    for (int i = 0; i < 2; ++i)
#pragma unroll
      for (int r = 0; r < 8; ++r)
        Cb[(long)(row0 + 16 * i + r) * ldc + colb + 16 * j] = acc[i][j][r] + bias;
  }
}

// ---------------------------------------------------------------------------
extern "C" void kernel_launch(void* const* d_in, const int* in_sizes, int n_in,
                              void* d_out, int out_size, void* d_ws, size_t ws_size,
                              hipStream_t stream) {
  const float* x    = (const float*)d_in[0];
  const float* ctx  = (const float*)d_in[1];
  const unsigned char* mask = (const unsigned char*)d_in[2];
  const int*   task = (const int*)d_in[3];
  const float* Wq   = (const float*)d_in[4];
  const float* Wk   = (const float*)d_in[5];
  const float* Wv   = (const float*)d_in[6];
  const float* Ak   = (const float*)d_in[7];
  const float* Bk   = (const float*)d_in[8];
  const float* Av   = (const float*)d_in[9];
  const float* Bv   = (const float*)d_in[10];
  const float* Wo   = (const float*)d_in[11];
  const float* bo   = (const float*)d_in[12];
  float* out = (float*)d_out;

  char* ws = (char*)d_ws;
  const size_t SZ = (size_t)BB * NN * INNER * sizeof(unsigned short);  // 8 MiB
  unsigned short* qb = (unsigned short*)(ws);
  unsigned short* kb = (unsigned short*)(ws + SZ);
  unsigned short* vb = (unsigned short*)(ws + 2 * SZ);
  unsigned short* ob = (unsigned short*)(ws + 3 * SZ);
  float* WkE = (float*)(ws + 4 * SZ);
  float* WvE = (float*)(ws + 4 * SZ + (size_t)BB * INNER * DD * sizeof(float));

  dim3 blk(256);

  build_weff<<<dim3(INNER * DD / 256, BB, 2), blk, 0, stream>>>(
      Wk, Ak, Bk, Wv, Av, Bv, task, WkE, WvE);

  const long sXZ = (long)NN * DD;        // per-batch x/context stride
  const long sQZ = (long)NN * INNER;     // per-batch q/k/v stride (elements)
  const long sWZ = (long)INNER * DD;     // per-batch W_eff stride

  gemm_f32_bf16out<<<dim3(NN / 128, INNER / 128, BB), blk, 0, stream>>>(
      x, sXZ, Wq, 0L, qb, sQZ, DD, INNER);
  gemm_f32_bf16out<<<dim3(MM / 128, INNER / 128, BB), blk, 0, stream>>>(
      ctx, sXZ, WkE, sWZ, kb, sQZ, DD, INNER);
  gemm_f32_bf16out<<<dim3(MM / 128, INNER / 128, BB), blk, 0, stream>>>(
      ctx, sXZ, WvE, sWZ, vb, sQZ, DD, INNER);

  attn_kernel<<<dim3(NN / 128, HH, BB), blk, 0, stream>>>(qb, kb, vb, mask, ob);

  gemm_bf16_f32bias<<<dim3(NN / 128, DD / 128, BB), blk, 0, stream>>>(
      ob, sQZ, Wo, out, (long)NN * DD, bo, INNER, DD);
}
